// GNMFeatureExtractor_29317446763029
// MI455X (gfx1250) — matrix operand order repaired and checked by reading, not compiled, not run on hardware
//
#include <hip/hip_runtime.h>
#include <math.h>

#define NN   1024
#define DIMF 32
#define KC   4
#define NS_ITERS 30

typedef __attribute__((ext_vector_type(16))) _Float16 v16h;
typedef __attribute__((ext_vector_type(8)))  float    v8f;
typedef __attribute__((ext_vector_type(2)))  float    v2f;
typedef __attribute__((ext_vector_type(4)))  int      v4i;

// ---------------------------------------------------------------------------
// Feature detection for gfx1250-specific builtins
// ---------------------------------------------------------------------------
#if __has_builtin(__builtin_amdgcn_wmma_f32_16x16x4_f32)
#define HAVE_WMMA_F32 1
#else
#define HAVE_WMMA_F32 0
#endif

#if __has_builtin(__builtin_amdgcn_global_load_async_to_lds_b128)
#define HAVE_ASYNC_LDS 1
#else
#define HAVE_ASYNC_LDS 0
#endif

#if HAVE_ASYNC_LDS
#if __has_builtin(__builtin_amdgcn_s_wait_asynccnt)
#define WAIT_ASYNC(n) __builtin_amdgcn_s_wait_asynccnt(n)
#else
#define WAIT_ASYNC(n) asm volatile("s_wait_asynccnt " #n ::: "memory")
#endif
#else
#define WAIT_ASYNC(n) ((void)0)
#endif

#define AS1 __attribute__((address_space(1)))
#define AS3 __attribute__((address_space(3)))
typedef AS1 v4i* gptr_v4i;
typedef AS3 v4i* lptr_v4i;

// 16-byte global -> LDS copy; async (ASYNCcnt) when available.
__device__ __forceinline__ void cp_async16(const float* g, float* l) {
#if HAVE_ASYNC_LDS
  __builtin_amdgcn_global_load_async_to_lds_b128((gptr_v4i)g, (lptr_v4i)l, 0, 0);
#else
  *(float4*)l = *(const float4*)g;
#endif
}

// ---------------------------------------------------------------------------
// WMMA f16 fragment helpers (CDNA5 16x16x32 layouts, wave32)
// A-matrix 16x32 f16: lane holds M = lane%16; half-slot h maps to
//   K = h + (h<8?0:8) + (lane>=16 ? 8 : 0)
// B-matrix 32x16 f16: lane holds N = lane%16; half-slot h maps to
//   K = h + (h<8?0:8) + (lane>=16 ? 8 : 0)
// C/D 16x16 f32: lane holds N = lane%16; reg v maps to M = v + 8*(lane>=16)
// ---------------------------------------------------------------------------
__device__ __forceinline__ int a_kmap(int h, int hf) {
  return h + ((h < 8) ? 0 : 8) + (hf ? 8 : 0);
}

__device__ __forceinline__ v16h load_w_frag(const float* __restrict__ W, int fanout,
                                            int ncol0, int lm, int hf) {
  const int n  = lm + ncol0;
  const int kb = hf * 16;
  v16h b;
#pragma unroll
  for (int h = 0; h < 16; ++h) {
    float w = (n < fanout) ? W[a_kmap(h, hf) * fanout + n] : 0.0f;
    b[h] = (_Float16)w;
  }
  return b;
}

// 3-layer MLP (32->32 relu ->32 relu ->4) on 16 pairs at once via WMMA.
__device__ __forceinline__ v8f mlp3_wmma(float x, float mu0, float dmu, float inv_sig,
                                         const float* __restrict__ W0, const float* __restrict__ B0,
                                         const float* __restrict__ W1, const float* __restrict__ B1,
                                         const float* __restrict__ W2, const float* __restrict__ B2,
                                         float (*buf)[DIMF], int lm, int hf) {
  const v8f zero = {};
  v16h a;
#pragma unroll
  for (int h = 0; h < 16; ++h) {
    int k = a_kmap(h, hf);
    float z = (x - (mu0 + dmu * (float)k)) * inv_sig;
    a[h] = (_Float16)__expf(-0.5f * z * z);
  }
  // ----- layer 1 -----
  v16h Ba = load_w_frag(W0, DIMF, 0, lm, hf);
  v16h Bb = load_w_frag(W0, DIMF, 16, lm, hf);
  v8f d0 = __builtin_amdgcn_wmma_f32_16x16x32_f16(false, a, false, Ba, (short)0, zero, false, false);
  v8f d1 = __builtin_amdgcn_wmma_f32_16x16x32_f16(false, a, false, Bb, (short)0, zero, false, false);
  float bn0 = B0[lm], bn1 = B0[lm + 16];
  __syncthreads();
#pragma unroll
  for (int v = 0; v < 8; ++v) {
    int m = v + hf * 8;
    buf[m][lm]      = fmaxf(d0[v] + bn0, 0.0f);
    buf[m][lm + 16] = fmaxf(d1[v] + bn1, 0.0f);
  }
  __syncthreads();
#pragma unroll
  for (int h = 0; h < 16; ++h) a[h] = (_Float16)buf[lm][a_kmap(h, hf)];
  // ----- layer 2 -----
  Ba = load_w_frag(W1, DIMF, 0, lm, hf);
  Bb = load_w_frag(W1, DIMF, 16, lm, hf);
  d0 = __builtin_amdgcn_wmma_f32_16x16x32_f16(false, a, false, Ba, (short)0, zero, false, false);
  d1 = __builtin_amdgcn_wmma_f32_16x16x32_f16(false, a, false, Bb, (short)0, zero, false, false);
  bn0 = B1[lm]; bn1 = B1[lm + 16];
  __syncthreads();
#pragma unroll
  for (int v = 0; v < 8; ++v) {
    int m = v + hf * 8;
    buf[m][lm]      = fmaxf(d0[v] + bn0, 0.0f);
    buf[m][lm + 16] = fmaxf(d1[v] + bn1, 0.0f);
  }
  __syncthreads();
#pragma unroll
  for (int h = 0; h < 16; ++h) a[h] = (_Float16)buf[lm][a_kmap(h, hf)];
  // ----- layer 3 (32 -> 4, B zero-padded to 16 columns) -----
  Ba = load_w_frag(W2, KC, 0, lm, hf);
  v8f dd = __builtin_amdgcn_wmma_f32_16x16x32_f16(false, a, false, Ba, (short)0, zero, false, false);
  float bo = (lm < KC) ? B2[lm] : 0.0f;
#pragma unroll
  for (int v = 0; v < 8; ++v) dd[v] += bo;
  return dd;
}

// ---------------------------------------------------------------------------
// Kernel 1: build A (KC, N, N).  One wave per (i, 16-wide j-tile); the 16
// pairs are the WMMA M dimension.
// ---------------------------------------------------------------------------
__global__ __launch_bounds__(256)
void pair_feature_kernel(const float* __restrict__ coords, const float* __restrict__ frames,
                         const float* __restrict__ w10, const float* __restrict__ b10,
                         const float* __restrict__ w11, const float* __restrict__ b11,
                         const float* __restrict__ w12, const float* __restrict__ b12,
                         const float* __restrict__ w20, const float* __restrict__ b20,
                         const float* __restrict__ w21, const float* __restrict__ b21,
                         const float* __restrict__ w22, const float* __restrict__ b22,
                         float* __restrict__ Amat)
{
  __shared__ float stage[8][16][DIMF];
  const int wave = threadIdx.x >> 5;
  const int lane = threadIdx.x & 31;
  const int hf   = lane >> 4;
  const int lm   = lane & 15;
  const int gw = blockIdx.x * 8 + wave;
  const int i  = gw / (NN / 16);
  const int j0 = (gw % (NN / 16)) * 16;
  const int j  = j0 + lm;              // this lane's pair (both halves compute m=lm)

  float dx = coords[i * 3 + 0] - coords[j * 3 + 0];
  float dy = coords[i * 3 + 1] - coords[j * 3 + 1];
  float dz = coords[i * 3 + 2] - coords[j * 3 + 2];
  float dist = sqrtf(dx * dx + dy * dy + dz * dz + 1e-12f);
  float score = 0.0f;
#pragma unroll
  for (int t = 0; t < 9; ++t) score += frames[i * 9 + t] * frames[j * 9 + t];

  v8f e1 = mlp3_wmma(dist,  0.0f, 20.0f / 31.0f, 31.0f / 20.0f,
                     w10, b10, w11, b11, w12, b12, stage[wave], lm, hf);
  v8f e2 = mlp3_wmma(score, -3.0f, 6.0f / 31.0f, 31.0f / 6.0f,
                     w20, b20, w21, b21, w22, b22, stage[wave], lm, hf);

  if (lm < KC) {
    const int c = lm;
#pragma unroll
    for (int v = 0; v < 8; ++v) {
      int m = v + hf * 8;
      Amat[((size_t)c * NN + i) * NN + (j0 + m)] = e1[v] * e2[v];
    }
  }
}

// ---------------------------------------------------------------------------
// Kernel 2: deg[c][i] = sum_j A[c][i][j]
// ---------------------------------------------------------------------------
__global__ __launch_bounds__(256)
void row_sum_kernel(const float* __restrict__ Amat, float* __restrict__ deg)
{
  __shared__ float red[256];
  const int i = blockIdx.x, c = blockIdx.y;
  const float* row = Amat + ((size_t)c * NN + i) * NN;
  float s = 0.0f;
  for (int j = threadIdx.x; j < NN; j += 256) s += row[j];
  red[threadIdx.x] = s; __syncthreads();
  for (int st = 128; st > 0; st >>= 1) {
    if (threadIdx.x < st) red[threadIdx.x] += red[threadIdx.x + st];
    __syncthreads();
  }
  if (threadIdx.x == 0) deg[c * NN + i] = red[0];
}

// ---------------------------------------------------------------------------
// Kernel 3: M = deg*I - A + 1/N (regularized Laplacian), plus abs row sums
// for the Gershgorin spectral bound.
// ---------------------------------------------------------------------------
__global__ __launch_bounds__(256)
void assemble_kernel(const float* __restrict__ Amat, const float* __restrict__ deg,
                     float* __restrict__ Mmat, float* __restrict__ rowabs)
{
  __shared__ float red[256];
  const int i = blockIdx.x, c = blockIdx.y;
  const float invN = 1.0f / (float)NN;
  const float d = deg[c * NN + i];
  const float* arow = Amat + ((size_t)c * NN + i) * NN;
  float* mrow = Mmat + ((size_t)c * NN + i) * NN;
  float s = 0.0f;
  for (int j = threadIdx.x; j < NN; j += 256) {
    float m = ((j == i) ? d : 0.0f) - arow[j] + invN;
    mrow[j] = m;
    s += fabsf(m);
  }
  red[threadIdx.x] = s; __syncthreads();
  for (int st = 128; st > 0; st >>= 1) {
    if (threadIdx.x < st) red[threadIdx.x] += red[threadIdx.x + st];
    __syncthreads();
  }
  if (threadIdx.x == 0) rowabs[c * NN + i] = red[0];
}

__global__ __launch_bounds__(256)
void alpha_kernel(const float* __restrict__ rowabs, float* __restrict__ alpha)
{
  __shared__ float red[256];
  const int c = blockIdx.x;
  float m = 0.0f;
  for (int i = threadIdx.x; i < NN; i += 256) m = fmaxf(m, rowabs[c * NN + i]);
  red[threadIdx.x] = m; __syncthreads();
  for (int st = 128; st > 0; st >>= 1) {
    if (threadIdx.x < st) red[threadIdx.x] = fmaxf(red[threadIdx.x], red[threadIdx.x + st]);
    __syncthreads();
  }
  // X0 = alpha*M with alpha = 1/bound^2 guarantees Newton-Schulz convergence
  // for symmetric (possibly indefinite) M.
  if (threadIdx.x == 0) alpha[c] = 1.0f / (red[0] * red[0] + 1e-30f);
}

__global__ __launch_bounds__(256)
void init_x_kernel(const float* __restrict__ Mmat, const float* __restrict__ alpha,
                   float* __restrict__ X)
{
  const size_t idx = (size_t)blockIdx.x * 256 + threadIdx.x;   // < KC*NN*NN
  const int c = (int)(idx >> 20);                              // NN*NN == 2^20
  X[idx] = alpha[c] * Mmat[idx];
}

// ---------------------------------------------------------------------------
// Newton-Schulz GEMM, fp32 WMMA (V_WMMA_F32_16X16X4_F32) with double-buffered
// async global->LDS staging (ASYNCcnt).
// mode 0: C = A*B          mode 1: C = 2*E - A*B   (E = current X)
// Block 256 thr = 8 waves; block tile 128x128; K panel 16; wave tile 64x32.
// LDS: A panel 2*128*20*4 = 20.0 KB (row stride 80B: 16B-aligned b128 stores,
//      conflict-free b64 fragment reads), B panel 2*16*128*4 = 16.0 KB.
// ---------------------------------------------------------------------------
#define GKB 16                  // K panel depth
#define GKP 20                  // padded A-panel row stride (floats)

__global__ __launch_bounds__(256)
void ns_gemm_kernel(const float* __restrict__ A, const float* __restrict__ B,
                    const float* __restrict__ E, float* __restrict__ C, int mode)
{
  const size_t ch = (size_t)blockIdx.z * NN * NN;
  const float* Ac = A + ch;
  const float* Bc = B + ch;
  const float* Ec = E + ch;
  float* Cc = C + ch;

#if HAVE_WMMA_F32
  __shared__ float As[2][128][GKP];
  __shared__ float Bs[2][GKB][128];

  const int tid  = threadIdx.x;
  const int wave = tid >> 5;
  const int lane = tid & 31;
  const int hf   = lane >> 4;
  const int lm   = lane & 15;
  const int row_g = blockIdx.y * 128;            // block tile origin
  const int col_g = blockIdx.x * 128;
  const int rw = (wave & 1) * 64;                // wave tile origin in block
  const int cw = (wave >> 1) * 32;

  // per-stage copy assignments (uniform across threads; 4 async ops per lane)
  const int am0 = (tid)        >> 2, ak0 = (tid)        & 3;
  const int am1 = (tid + 256)  >> 2, ak1 = (tid + 256)  & 3;
  const int bk0 = (tid)        >> 5, bn0 = (tid)        & 31;
  const int bk1 = (tid + 256)  >> 5, bn1 = (tid + 256)  & 31;

  v8f acc[4][2];
  const v8f zero = {};
#pragma unroll
  for (int t = 0; t < 4; ++t)
#pragma unroll
    for (int u = 0; u < 2; ++u) acc[t][u] = zero;

#define ISSUE_STAGE(kt, b)                                                          \
  do {                                                                              \
    const int k0_ = (kt) * GKB;                                                     \
    cp_async16(Ac + (size_t)(row_g + am0) * NN + k0_ + ak0 * 4, &As[b][am0][ak0*4]);\
    cp_async16(Ac + (size_t)(row_g + am1) * NN + k0_ + ak1 * 4, &As[b][am1][ak1*4]);\
    cp_async16(Bc + (size_t)(k0_ + bk0) * NN + col_g + bn0 * 4, &Bs[b][bk0][bn0*4]);\
    cp_async16(Bc + (size_t)(k0_ + bk1) * NN + col_g + bn1 * 4, &Bs[b][bk1][bn1*4]);\
  } while (0)

  const int NT = NN / GKB;          // 64 stages
  ISSUE_STAGE(0, 0);
  for (int kt = 0; kt < NT; ++kt) {
    const int cur = kt & 1;
    if (kt + 1 < NT) {
      ISSUE_STAGE(kt + 1, cur ^ 1);
      WAIT_ASYNC(4);                 // in-order: stage kt's 4 copies complete
    } else {
      WAIT_ASYNC(0);
    }
    __syncthreads();
#pragma unroll
    for (int ks = 0; ks < GKB; ks += 4) {
      const int ka = ks + 2 * hf;
      v2f af[4], bf[2];
#pragma unroll
      for (int t = 0; t < 4; ++t)
        af[t] = *(const v2f*)&As[cur][rw + t * 16 + lm][ka];
#pragma unroll
      for (int u = 0; u < 2; ++u) {
        const int cc = cw + u * 16 + lm;
        bf[u].x = Bs[cur][ka][cc];
        bf[u].y = Bs[cur][ka + 1][cc];
      }
#pragma unroll
      for (int t = 0; t < 4; ++t)
#pragma unroll
        for (int u = 0; u < 2; ++u)
          acc[t][u] = __builtin_amdgcn_wmma_f32_16x16x4_f32(
              false, af[t], false, bf[u], (short)0, acc[t][u], false, false);
    }
    __syncthreads();                 // all waves done with `cur` before reuse
  }
#undef ISSUE_STAGE

#pragma unroll
  for (int t = 0; t < 4; ++t)
#pragma unroll
    for (int u = 0; u < 2; ++u) {
#pragma unroll
      for (int v = 0; v < 8; ++v) {
        const int r  = row_g + rw + t * 16 + v + 8 * hf;
        const int cc = col_g + cw + u * 16 + lm;
        const size_t off = (size_t)r * NN + cc;
        float val = acc[t][u][v];
        if (mode) val = 2.0f * Ec[off] - val;
        Cc[off] = val;
      }
    }
#else
  // Scalar fallback (compiles if the f32 WMMA builtin is unavailable).
  const int tid = threadIdx.x;
  const int r  = blockIdx.y * 128 + (tid >> 1);
  const int c0 = blockIdx.x * 128 + (tid & 1) * 64;
  for (int cc = 0; cc < 64; ++cc) {
    float acc = 0.0f;
    for (int k = 0; k < NN; ++k)
      acc = fmaf(Ac[(size_t)r * NN + k], Bc[(size_t)k * NN + c0 + cc], acc);
    const size_t off = (size_t)r * NN + c0 + cc;
    Cc[off] = mode ? (2.0f * Ec[off] - acc) : acc;
  }
#endif
}

// ---------------------------------------------------------------------------
// Kernel: out[i][c] = X[c][i][i] - 1/N    (pinv(L) diag via (L+J/N)^-1 - J/N)
// ---------------------------------------------------------------------------
__global__ __launch_bounds__(256)
void diag_kernel(const float* __restrict__ X, float* __restrict__ out)
{
  const int idx = blockIdx.x * 256 + threadIdx.x;  // < NN*KC
  const int i = idx >> 2, c = idx & 3;
  out[idx] = X[(size_t)c * NN * NN + (size_t)i * NN + i] - 1.0f / (float)NN;
}

// ---------------------------------------------------------------------------
extern "C" void kernel_launch(void* const* d_in, const int* in_sizes, int n_in,
                              void* d_out, int out_size, void* d_ws, size_t ws_size,
                              hipStream_t stream)
{
  (void)in_sizes; (void)n_in; (void)out_size; (void)ws_size;
  const float* coords = (const float*)d_in[0];
  const float* frames = (const float*)d_in[1];
  const float* w10 = (const float*)d_in[2];  const float* b10 = (const float*)d_in[3];
  const float* w11 = (const float*)d_in[4];  const float* b11 = (const float*)d_in[5];
  const float* w12 = (const float*)d_in[6];  const float* b12 = (const float*)d_in[7];
  const float* w20 = (const float*)d_in[8];  const float* b20 = (const float*)d_in[9];
  const float* w21 = (const float*)d_in[10]; const float* b21 = (const float*)d_in[11];
  const float* w22 = (const float*)d_in[12]; const float* b22 = (const float*)d_in[13];

  char* ws = (char*)d_ws;
  const size_t MATB = (size_t)KC * NN * NN * sizeof(float);   // 16 MB per matrix set
  float* Amat   = (float*)(ws);              // also reused as T after assembly
  float* Mmat   = (float*)(ws + MATB);
  float* X1     = (float*)(ws + 2 * MATB);
  float* X2     = (float*)(ws + 3 * MATB);
  float* Tbuf   = Amat;
  float* deg    = (float*)(ws + 4 * MATB);
  float* rowabs = deg + KC * NN;
  float* alpha  = rowabs + KC * NN;

  // 1) A = emb1 * emb2 over all pairs (f16 WMMA MLPs)
  pair_feature_kernel<<<(NN * (NN / 16)) / 8, 256, 0, stream>>>(
      coords, frames, w10, b10, w11, b11, w12, b12,
      w20, b20, w21, b21, w22, b22, Amat);
  // 2) degrees, M = L + J/N, spectral bound, X0 = alpha*M
  row_sum_kernel<<<dim3(NN, KC), 256, 0, stream>>>(Amat, deg);
  assemble_kernel<<<dim3(NN, KC), 256, 0, stream>>>(Amat, deg, Mmat, rowabs);
  alpha_kernel<<<KC, 256, 0, stream>>>(rowabs, alpha);
  init_x_kernel<<<(KC * NN * NN) / 256, 256, 0, stream>>>(Mmat, alpha, X1);

  // 3) Newton-Schulz: X <- 2X - X*M*X   (fp32 WMMA GEMMs, batched over KC)
  const dim3 gg(NN / 128, NN / 128, KC);
  float* X = X1; float* Xo = X2;
  for (int it = 0; it < NS_ITERS; ++it) {
    ns_gemm_kernel<<<gg, 256, 0, stream>>>(Mmat, X, X, Tbuf, 0);   // T = M*X
    ns_gemm_kernel<<<gg, 256, 0, stream>>>(X, Tbuf, X, Xo, 1);     // X' = 2X - X*T
    float* tmp = X; X = Xo; Xo = tmp;
  }

  // 4) variances = diag(M^-1) - 1/N
  diag_kernel<<<(NN * KC) / 256, 256, 0, stream>>>(X, (float*)d_out);
}
